// SemisupKernelClassifier_3375844295427
// MI455X (gfx1250) — matrix operand order repaired and checked by reading, not compiled, not run on hardware
//
#include <hip/hip_runtime.h>
#include <hip/hip_bf16.h>
#include <math.h>

typedef __attribute__((ext_vector_type(16))) _Float16 v16h;
typedef __attribute__((ext_vector_type(8)))  _Float16 v8h;
typedef __attribute__((ext_vector_type(8)))  float    v8f;

#define NX 4096
#define DIM 64
#define NL 2048
#define NU 8192
#define MA (NL + NU)            /* 10240 */
#define NC 10
#define JTILES (MA / 16)        /* 640 */
#define JSPLIT 8
#define JTPB (JTILES / JSPLIT)  /* 80 */
#define INV_SQRT_2PI 0.39894228040143267794f

/* ---- workspace layout (bytes) ---- */
#define WS_XHI   0u          /* 4096*64 f16  = 512KB  */
#define WS_XLO   524288u
#define WS_AHI   1048576u    /* 10240*64 f16 = 1.25MB */
#define WS_ALO   2359296u
#define WS_X2    3670016u    /* 4096 f32  */
#define WS_A2    3686400u    /* 10240 f32 */
#define WS_PDY   3727360u    /* 10240*10 f32 */
#define WS_ACCXY 4136960u    /* 4096*10 f32 */
#define WS_ACCPX 4300800u    /* 4096 f32 (contiguous with ACCXY) */

static __device__ __forceinline__ v16h cat8(v8h a, v8h b) {
  return __builtin_shufflevector(a, b, 0,1,2,3,4,5,6,7,8,9,10,11,12,13,14,15);
}
/* A-matrix 16x32 f16 layout: lanes 0-15 rows M=0..15 hold K = {0..7,16..23};
   lanes 16-31 hold K = {8..15,24..31}.  Two 16B loads per lane. */
static __device__ __forceinline__ v16h load_tileA(const _Float16* row, int ks, int half_id) {
  const _Float16* p = row + ks + half_id * 8;
  v8h a = *(const v8h*)p;
  v8h b = *(const v8h*)(p + 16);
  return cat8(a, b);
}
/* B-matrix 32x16 f16 layout: lane holds column n=lane%16; lanes 0-15 hold
   K = {0..7,16..23}, lanes 16-31 hold K = {8..15,24..31}.  Two 16B loads per lane. */
static __device__ __forceinline__ v16h load_tileB(const _Float16* row, int ks, int half_id) {
  const _Float16* p = row + ks + half_id * 8;
  v8h a = *(const v8h*)p;
  v8h b = *(const v8h*)(p + 16);
  return cat8(a, b);
}
static __device__ __forceinline__ v8f wmma_f16(v16h a, v16h b, v8f c) {
  return __builtin_amdgcn_wmma_f32_16x16x32_f16(false, a, false, b, (short)0, c, false, false);
}

/* ---------- prologue kernels ---------- */
__global__ void split_f16(const float* __restrict__ src, _Float16* __restrict__ hi,
                          _Float16* __restrict__ lo, int n) {
  int i = blockIdx.x * blockDim.x + threadIdx.x;
  if (i >= n) return;
  float v = src[i];
  _Float16 h = (_Float16)v;
  hi[i] = h;
  lo[i] = (_Float16)(v - (float)h);
}

__global__ void row_sumsq(const float* __restrict__ src, float* __restrict__ dst, int rows) {
  int i = blockIdx.x * blockDim.x + threadIdx.x;
  if (i >= rows) return;
  const float* r = src + (size_t)i * DIM;
  float s = 0.f;
#pragma unroll
  for (int k = 0; k < DIM; ++k) s = fmaf(r[k], r[k], s);
  dst[i] = s;
}

__global__ void pdy_onehot(const int* __restrict__ y, float* __restrict__ pdy) {
  int i = blockIdx.x * blockDim.x + threadIdx.x;
  if (i >= NL) return;
  int yi = y[i];
#pragma unroll
  for (int c = 0; c < NC; ++c) pdy[i * NC + c] = (c == yi) ? 1.f : 0.f;
}

__global__ void pdy_softmax(const float* __restrict__ s, float* __restrict__ pdy) {
  int i = blockIdx.x * blockDim.x + threadIdx.x;
  if (i >= NU) return;
  float v[NC];
  float m = -INFINITY;
#pragma unroll
  for (int c = 0; c < NC; ++c) { v[c] = s[i * NC + c]; m = fmaxf(m, v[c]); }
  float sum = 0.f;
#pragma unroll
  for (int c = 0; c < NC; ++c) { v[c] = __expf(v[c] - m); sum += v[c]; }
  float inv = 1.f / sum;
#pragma unroll
  for (int c = 0; c < NC; ++c) pdy[(NL + i) * NC + c] = v[c] * inv;
}

__global__ void zero_f32(float* __restrict__ p, int n) {
  int i = blockIdx.x * blockDim.x + threadIdx.x;
  if (i < n) p[i] = 0.f;
}

/* ---------- fused WMMA main kernel ----------
   One wave per (16-row i-tile, j-chunk).  A operands (x rows, hi+lo, two
   K-slices) are loop-invariant registers; per j-tile we issue 6 WMMAs
   (hi*hi + hi*lo + lo*hi over K=64), then exp + 10-class FMA update. */
__global__ __launch_bounds__(32) void semisup_main(
    const _Float16* __restrict__ xh, const _Float16* __restrict__ xl,
    const _Float16* __restrict__ ah, const _Float16* __restrict__ al,
    const float* __restrict__ x2,  const float* __restrict__ a2,
    const float* __restrict__ pdy,
    float* __restrict__ acc_xy, float* __restrict__ acc_px) {
  const int lane    = threadIdx.x;
  const int lane16  = lane & 15;
  const int half_id = lane >> 4;
  const int itile   = blockIdx.x;
  const int jchunk  = blockIdx.y;

  const _Float16* xrow  = xh + (size_t)(itile * 16 + lane16) * DIM;
  const _Float16* xrowl = xl + (size_t)(itile * 16 + lane16) * DIM;
  const v16h Ah0 = load_tileA(xrow,  0, half_id);
  const v16h Ah1 = load_tileA(xrow, 32, half_id);
  const v16h Al0 = load_tileA(xrowl, 0, half_id);
  const v16h Al1 = load_tileA(xrowl,32, half_id);

  float x2v[8];
#pragma unroll
  for (int r = 0; r < 8; ++r) x2v[r] = x2[itile * 16 + half_id * 8 + r];

  float px[8];
  float xy[8][NC];
#pragma unroll
  for (int r = 0; r < 8; ++r) {
    px[r] = 0.f;
#pragma unroll
    for (int c = 0; c < NC; ++c) xy[r][c] = 0.f;
  }

#pragma unroll 1
  for (int jt = 0; jt < JTPB; ++jt) {
    const int jbase = (jchunk * JTPB + jt) * 16;
    const int j = jbase + lane16;               /* this lane's x_a row / C column */
    const _Float16* arow  = ah + (size_t)j * DIM;
    const _Float16* arowl = al + (size_t)j * DIM;
    const v16h Bh0 = load_tileB(arow,  0, half_id);
    const v16h Bh1 = load_tileB(arow, 32, half_id);
    const v16h Bl0 = load_tileB(arowl, 0, half_id);
    const v16h Bl1 = load_tileB(arowl,32, half_id);

    v8f c = {};
    c = wmma_f16(Ah0, Bh0, c);
    c = wmma_f16(Ah1, Bh1, c);
    c = wmma_f16(Ah0, Bl0, c);
    c = wmma_f16(Ah1, Bl1, c);
    c = wmma_f16(Al0, Bh0, c);
    c = wmma_f16(Al1, Bh1, c);

    const float a2j = a2[j];
    const float* wrow = pdy + (size_t)j * NC;
    float w[NC];
#pragma unroll
    for (int cc = 0; cc < NC; ++cc) w[cc] = wrow[cc];

#pragma unroll
    for (int r = 0; r < 8; ++r) {
      float d2 = x2v[r] + a2j - 2.0f * c[r];
      float p  = __expf(-0.5f * d2) * INV_SQRT_2PI;
      px[r] += p;
#pragma unroll
      for (int cc = 0; cc < NC; ++cc) xy[r][cc] = fmaf(p, w[cc], xy[r][cc]);
    }
  }

  /* reduce across the 16 lanes of each half (different j columns) */
#pragma unroll
  for (int mask = 1; mask <= 8; mask <<= 1) {
#pragma unroll
    for (int r = 0; r < 8; ++r) {
      px[r] += __shfl_xor(px[r], mask, 32);
#pragma unroll
      for (int cc = 0; cc < NC; ++cc) xy[r][cc] += __shfl_xor(xy[r][cc], mask, 32);
    }
  }

  if (lane16 == 0) {   /* lanes 0 (rows 0-7) and 16 (rows 8-15) commit */
#pragma unroll
    for (int r = 0; r < 8; ++r) {
      int row = itile * 16 + half_id * 8 + r;
      atomicAdd(&acc_px[row], px[r]);
#pragma unroll
      for (int cc = 0; cc < NC; ++cc)
        atomicAdd(&acc_xy[row * NC + cc], xy[r][cc]);
    }
  }
}

__global__ void finalize(const float* __restrict__ xy, const float* __restrict__ px,
                         float* __restrict__ out) {
  int i = blockIdx.x * blockDim.x + threadIdx.x;
  if (i >= NX * NC) return;
  out[i] = xy[i] / px[i / NC];
}

extern "C" void kernel_launch(void* const* d_in, const int* in_sizes, int n_in,
                              void* d_out, int out_size, void* d_ws, size_t ws_size,
                              hipStream_t stream) {
  const float* x  = (const float*)d_in[0];
  const float* xl = (const float*)d_in[1];
  const float* xu = (const float*)d_in[2];
  const float* su = (const float*)d_in[3];
  const int*   yl = (const int*)d_in[4];
  float* out = (float*)d_out;

  char* ws = (char*)d_ws;
  _Float16* x_hi = (_Float16*)(ws + WS_XHI);
  _Float16* x_lo = (_Float16*)(ws + WS_XLO);
  _Float16* a_hi = (_Float16*)(ws + WS_AHI);
  _Float16* a_lo = (_Float16*)(ws + WS_ALO);
  float* x2    = (float*)(ws + WS_X2);
  float* a2    = (float*)(ws + WS_A2);
  float* pdy   = (float*)(ws + WS_PDY);
  float* accxy = (float*)(ws + WS_ACCXY);
  /* acc_px contiguous after acc_xy */
  float* accpx = (float*)(ws + WS_ACCPX);

  /* prologue: f16 hi/lo copies, row norms, p_dy, zeroed staging */
  split_f16<<<(NX * DIM + 255) / 256, 256, 0, stream>>>(x,  x_hi, x_lo, NX * DIM);
  split_f16<<<(NL * DIM + 255) / 256, 256, 0, stream>>>(xl, a_hi, a_lo, NL * DIM);
  split_f16<<<(NU * DIM + 255) / 256, 256, 0, stream>>>(xu, a_hi + NL * DIM, a_lo + NL * DIM, NU * DIM);
  row_sumsq<<<(NX + 255) / 256, 256, 0, stream>>>(x,  x2, NX);
  row_sumsq<<<(NL + 255) / 256, 256, 0, stream>>>(xl, a2, NL);
  row_sumsq<<<(NU + 255) / 256, 256, 0, stream>>>(xu, a2 + NL, NU);
  pdy_onehot<<<(NL + 255) / 256, 256, 0, stream>>>(yl, pdy);
  pdy_softmax<<<(NU + 255) / 256, 256, 0, stream>>>(su, pdy);
  zero_f32<<<(NX * NC + NX + 255) / 256, 256, 0, stream>>>(accxy, NX * NC + NX);

  /* fused kernel-matrix + class GEMM + row-sum */
  dim3 grid(NX / 16, JSPLIT);
  semisup_main<<<grid, 32, 0, stream>>>(x_hi, x_lo, a_hi, a_lo, x2, a2, pdy, accxy, accpx);

  finalize<<<(NX * NC + 255) / 256, 256, 0, stream>>>(accxy, accpx, out);
}
